// Model_90675349553336
// MI455X (gfx1250) — matrix operand order repaired and checked by reading, not compiled, not run on hardware
//
#include <hip/hip_runtime.h>
#include <math.h>

// ---------------------------------------------------------------------------
// Types for CDNA5 WMMA
// ---------------------------------------------------------------------------
typedef __attribute__((ext_vector_type(16))) __bf16 v16bf;
typedef __attribute__((ext_vector_type(8)))  float  v8f;

union Frag { uint4 u[2]; v16bf v; };

// Problem constants (B=2048, L=2048, C=16, F=1024, H=4096, P=88)
#define NB   2048
#define NL   2048
#define NC   16
#define NF   1024
#define NH   4096
#define NP   88
#define N2F  2048   // 2*F (K of layer1, N of layer2)
#define N2H  8192   // 2*H (N of layer1, K of layer2)
#define SAS  40     // padded LDS row stride (32 + 8 bf16) -> conflict-free frag reads

// ---------------------------------------------------------------------------
// CDNA5 async global->LDS copy (ASYNCcnt-tracked DMA path, ISA 15.18.3)
// vdst = 32-bit LDS byte address, vaddr = 64-bit global address
// ---------------------------------------------------------------------------
__device__ __forceinline__ void async_copy_b128(unsigned lds_addr, const void* gaddr) {
  asm volatile("global_load_async_to_lds_b128 %0, %1, off"
               :: "v"(lds_addr), "v"(gaddr)
               : "memory");
}
__device__ __forceinline__ void wait_async0() {
  asm volatile("s_wait_asynccnt 0x0" ::: "memory");
}
// async loads complete in order -> waiting to <=4 outstanding guarantees the
// oldest tile's 4 ops are done while the newest 4 stay in flight.
__device__ __forceinline__ void wait_async4() {
  asm volatile("s_wait_asynccnt 0x4" ::: "memory");
}

// ---------------------------------------------------------------------------
// Prep: fold complex structure into pre-transposed bf16 weights  Bt[n][k]
// layer1: K=2F, N=2H ; o1r cols use [w10 ; -w11], o1i cols use [w11 ; w10]
// ---------------------------------------------------------------------------
__global__ __launch_bounds__(256) void prep_w1t(const float* __restrict__ w1,
                                                __bf16* __restrict__ W1t) {
  int idx = blockIdx.x * 256 + threadIdx.x;      // 8192*2048 elements
  int n = idx >> 11;                             // / 2048
  int k = idx & 2047;
  const int FH = NF * NH;
  float v;
  if (n < NH) {
    v = (k < NF) ? w1[k * NH + n] : -w1[FH + (k - NF) * NH + n];
  } else {
    int nn = n - NH;
    v = (k < NF) ? w1[FH + k * NH + nn] : w1[(k - NF) * NH + nn];
  }
  W1t[idx] = (__bf16)v;
}

// layer2: K=2H, N=2F ; o2r cols use [w20 ; -w21], o2i cols use [w21 ; w20]
__global__ __launch_bounds__(256) void prep_w2t(const float* __restrict__ w2,
                                                __bf16* __restrict__ W2t) {
  int idx = blockIdx.x * 256 + threadIdx.x;      // 2048*8192 elements
  int n = idx >> 13;                             // / 8192
  int k = idx & 8191;
  const int HF = NH * NF;
  float v;
  if (n < NF) {
    v = (k < NH) ? w2[k * NF + n] : -w2[HF + (k - NH) * NF + n];
  } else {
    int nn = n - NF;
    v = (k < NH) ? w2[HF + k * NF + nn] : w2[(k - NH) * NF + nn];
  }
  W2t[idx] = (__bf16)v;
}

// ---------------------------------------------------------------------------
// Fused start_fc + real FFT(2048) (Stockham radix-2, ortho norm, drop DC)
// act1[b][0..1023]  = Re(rfft(v)[1..1024]) / sqrt(2048)   (bf16)
// act1[b][1024..]   = Im(rfft(v)[1..1024]) / sqrt(2048)
// ---------------------------------------------------------------------------
__global__ __launch_bounds__(256) void start_fft_kernel(
    const float* __restrict__ x, const float* __restrict__ w_start,
    const float* __restrict__ b_start, __bf16* __restrict__ act1) {
  __shared__ float2 bufA[NL];
  __shared__ float2 bufB[NL];
  const int t = threadIdx.x;
  const size_t b = blockIdx.x;

  float ws[NC];
#pragma unroll
  for (int c = 0; c < NC; ++c) ws[c] = w_start[c];
  const float bs = b_start[0];

  const float* xb = x + b * (size_t)NL * NC;
#pragma unroll
  for (int i = 0; i < 8; ++i) {
    int l = t + i * 256;
    const float4* p = (const float4*)(xb + (size_t)l * NC);
    float4 a0 = p[0], a1 = p[1], a2 = p[2], a3 = p[3];
    float v = bs;
    v += a0.x * ws[0]  + a0.y * ws[1]  + a0.z * ws[2]  + a0.w * ws[3];
    v += a1.x * ws[4]  + a1.y * ws[5]  + a1.z * ws[6]  + a1.w * ws[7];
    v += a2.x * ws[8]  + a2.y * ws[9]  + a2.z * ws[10] + a2.w * ws[11];
    v += a3.x * ws[12] + a3.y * ws[13] + a3.z * ws[14] + a3.w * ws[15];
    bufA[l] = make_float2(v, 0.f);
  }
  __syncthreads();

  float2* s = bufA;
  float2* d = bufB;
  for (int lg = 0; lg < 11; ++lg) {
    const int ns = 1 << lg;
#pragma unroll
    for (int i = 0; i < 4; ++i) {
      int j = t + i * 256;                  // 0..1023
      float2 u = s[j];
      float2 w = s[j + 1024];
      int r = j & (ns - 1);
      float ang = -3.14159265358979323846f * (float)r / (float)ns;
      float sn, cs;
      __sincosf(ang, &sn, &cs);
      float2 tw;
      tw.x = cs * w.x - sn * w.y;
      tw.y = cs * w.y + sn * w.x;
      int idxD = ((j >> lg) << (lg + 1)) + r;
      d[idxD]      = make_float2(u.x + tw.x, u.y + tw.y);
      d[idxD + ns] = make_float2(u.x - tw.x, u.y - tw.y);
    }
    __syncthreads();
    float2* tmp = s; s = d; d = tmp;
  }

  const float S = 0.02209708691207961f;     // 1/sqrt(2048), ortho norm
#pragma unroll
  for (int i = 0; i < 4; ++i) {
    int f = 1 + t + i * 256;                // bins 1..1024
    float2 v = s[f];
    act1[b * N2F + (f - 1)]      = (__bf16)(v.x * S);
    act1[b * N2F + NF + (f - 1)] = (__bf16)(v.y * S);
  }
}

// ---------------------------------------------------------------------------
// bf16 WMMA GEMM:  Out[M][N] = A[M][K] * Bt[N][K]^T + bias[N]
// EPI==0: relu -> bf16 output (layer1);  EPI==1: f32 output (layer2)
// Block: 256 thr (8 waves), tile 128x128x32. Wave: 4 m-tiles x 2 n-tiles.
// Triple-buffered LDS filled by CDNA5 async global->LDS DMA, issued two tiles
// ahead; `s_wait_asynccnt 4` retires only the oldest tile (async loads
// complete in order), so one tile's DMA is always in flight behind the WMMAs.
// One workgroup barrier per k-tile. Fragment layouts follow ISA 7.12.2.
// ---------------------------------------------------------------------------
template <int EPI>
__global__ __launch_bounds__(256) void gemm_bf16(
    const __bf16* __restrict__ A, const __bf16* __restrict__ Bt,
    const float* __restrict__ bias, void* __restrict__ Out,
    int M, int N, int K) {
  __shared__ __align__(16) __bf16 sA[3][128 * SAS];
  __shared__ __align__(16) __bf16 sB[3][128 * SAS];

  const int t     = threadIdx.x;
  const int lane  = t & 31;
  const int wave  = t >> 5;
  const int waveM = wave >> 2;      // 0..1  (64 rows each)
  const int waveN = wave & 3;       // 0..3  (32 cols each)
  const int bM = blockIdx.y * 128;
  const int bN = blockIdx.x * 128;

  // cooperative tile load mapping: thread -> (row, 16-elem half of 32 cols)
  const int lrow  = t >> 1;         // 0..127
  const int lhalf = (t & 1) * 16;   // 0 or 16
  const __bf16* gA = A  + (size_t)(bM + lrow) * K + lhalf;
  const __bf16* gB = Bt + (size_t)(bN + lrow) * K + lhalf;

  // LDS byte addresses for this thread's async destinations
  const unsigned bufStride = 128 * SAS * 2;  // bytes per buffer
  const unsigned ldsA0 = (unsigned)(size_t)(&sA[0][lrow * SAS + lhalf]);
  const unsigned ldsB0 = (unsigned)(size_t)(&sB[0][lrow * SAS + lhalf]);

  const v8f vzero = {0.f, 0.f, 0.f, 0.f, 0.f, 0.f, 0.f, 0.f};
  v8f acc[4][2];
#pragma unroll
  for (int i = 0; i < 4; ++i)
#pragma unroll
    for (int j = 0; j < 2; ++j) acc[i][j] = vzero;

  const int nk = K >> 5;   // >= 2 for all our shapes

  // prologue: async-DMA k-tiles 0 and 1 into buffers 0 and 1
  {
    async_copy_b128(ldsA0,      gA);
    async_copy_b128(ldsA0 + 16, gA + 8);
    async_copy_b128(ldsB0,      gB);
    async_copy_b128(ldsB0 + 16, gB + 8);
    if (nk > 1) {
      const __bf16* pA = gA + 32;
      const __bf16* pB = gB + 32;
      async_copy_b128(ldsA0 + bufStride,      pA);
      async_copy_b128(ldsA0 + bufStride + 16, pA + 8);
      async_copy_b128(ldsB0 + bufStride,      pB);
      async_copy_b128(ldsB0 + bufStride + 16, pB + 8);
      wait_async4();                 // tile 0 complete; tile 1 may fly
    } else {
      wait_async0();
    }
  }
  __syncthreads();

  const int mrow = lane & 15;
  const int half = lane >> 4;

  int cur = 0;                       // rotating buffer index (kb % 3)
  for (int kb = 0; kb < nk; ++kb) {
    // kick off async DMA of tile kb+2 into the buffer freed last iteration;
    // it overlaps with the WMMA work below.
    const bool issue = (kb + 2 < nk);
    if (issue) {
      int nxt2 = cur + 2; if (nxt2 >= 3) nxt2 -= 3;
      const unsigned off = (unsigned)nxt2 * bufStride;
      const __bf16* pA = gA + (size_t)(kb + 2) * 32;
      const __bf16* pB = gB + (size_t)(kb + 2) * 32;
      async_copy_b128(ldsA0 + off,      pA);
      async_copy_b128(ldsA0 + off + 16, pA + 8);
      async_copy_b128(ldsB0 + off,      pB);
      async_copy_b128(ldsB0 + off + 16, pB + 8);
    }

    // ---- fragment loads from LDS (ISA 7.12.2 layouts) ----
    Frag af[4], bfr[2];
#pragma unroll
    for (int i = 0; i < 4; ++i) {
      const __bf16* rp = &sA[cur][(waveM * 64 + i * 16 + mrow) * SAS];
      af[i].u[0] = *(const uint4*)(rp + half * 8);        // K = half*8 + 0..7
      af[i].u[1] = *(const uint4*)(rp + 16 + half * 8);   // K = 16+half*8+0..7
    }
#pragma unroll
    for (int j = 0; j < 2; ++j) {
      const __bf16* rp = &sB[cur][(waveN * 32 + j * 16 + mrow) * SAS];
      bfr[j].u[0] = *(const uint4*)(rp + half * 8);       // K = half*8 + 0..7
      bfr[j].u[1] = *(const uint4*)(rp + 16 + half * 8);  // K = 16+half*8+0..7
    }

#pragma unroll
    for (int i = 0; i < 4; ++i)
#pragma unroll
      for (int j = 0; j < 2; ++j)
        acc[i][j] = __builtin_amdgcn_wmma_f32_16x16x32_bf16(
            false, af[i].v, false, bfr[j].v, (short)0, acc[i][j], false, false);

    // retire the oldest in-flight tile (kb+1) for this wave, then one
    // workgroup barrier: afterwards every wave's DMA into that buffer is
    // visible and the buffer we just read may be overwritten next iteration.
    if (issue) wait_async4();
    else       wait_async0();
    __syncthreads();

    ++cur; if (cur >= 3) cur = 0;
  }

  // ---- epilogue: C/D layout -> lane (n = lane&15), VGPR r -> M = r + 8*(lane>>4)
#pragma unroll
  for (int i = 0; i < 4; ++i) {
#pragma unroll
    for (int j = 0; j < 2; ++j) {
      const int gN = bN + waveN * 32 + j * 16 + mrow;
      const float bv = bias[gN];
#pragma unroll
      for (int r = 0; r < 8; ++r) {
        const int gM = bM + waveM * 64 + i * 16 + half * 8 + r;
        float v = acc[i][j][r] + bv;
        if (EPI == 0) {
          v = v > 0.f ? v : 0.f;
          ((__bf16*)Out)[(size_t)gM * N + gN] = (__bf16)v;
        } else {
          ((float*)Out)[(size_t)gM * N + gN] = v;
        }
      }
    }
  }
}

// ---------------------------------------------------------------------------
// amp[b][f] = sqrt(o2r^2 + o2i^2)   (o2 layout: [b][0..1023]=re, [1024..]=im)
// ---------------------------------------------------------------------------
__global__ __launch_bounds__(256) void amp_kernel(const float* __restrict__ o2,
                                                  float* __restrict__ amp) {
  int idx = blockIdx.x * 256 + threadIdx.x;      // 2048*1024
  int b = idx >> 10, f = idx & 1023;
  float re = o2[(size_t)b * N2F + f];
  float im = o2[(size_t)b * N2F + NF + f];
  amp[idx] = sqrtf(re * re + im * im);
}

// ---------------------------------------------------------------------------
// raw[b][0..87]  = amp[b] @ w_gate ; raw[b][88..175] = amp[b] @ w_noise
// 16 batch rows / block, K staged in LDS chunks of 64.
// ---------------------------------------------------------------------------
__global__ __launch_bounds__(256) void router_kernel(
    const float* __restrict__ amp, const float* __restrict__ wg,
    const float* __restrict__ wn, float* __restrict__ raw) {
  __shared__ float sW[64 * 176];
  __shared__ float sAmp[16 * 64];
  const int t = threadIdx.x;
  const int b0 = blockIdx.x * 16;
  const int r = t >> 4, cg = t & 15;

  float acc[11];
#pragma unroll
  for (int j = 0; j < 11; ++j) acc[j] = 0.f;

  for (int kb = 0; kb < NF; kb += 64) {
    for (int idx = t; idx < 64 * 176; idx += 256) {
      int kk = idx / 176, p = idx % 176;
      sW[idx] = (p < NP) ? wg[(kb + kk) * NP + p] : wn[(kb + kk) * NP + (p - NP)];
    }
    for (int idx = t; idx < 16 * 64; idx += 256) {
      int rr = idx >> 6, kk = idx & 63;
      sAmp[idx] = amp[(size_t)(b0 + rr) * NF + kb + kk];
    }
    __syncthreads();
    for (int kk = 0; kk < 64; ++kk) {
      float a = sAmp[r * 64 + kk];
#pragma unroll
      for (int j = 0; j < 11; ++j) acc[j] += a * sW[kk * 176 + cg + 16 * j];
    }
    __syncthreads();
  }
#pragma unroll
  for (int j = 0; j < 11; ++j) raw[(b0 + r) * 176 + cg + 16 * j] = acc[j];
}

// ---------------------------------------------------------------------------
// Noisy top-3 gating: logits = clean + z*(softplus(noise)+eps); top-3 softmax
// scattered into gates[b][88]. One wave32 per batch row (3 candidates/lane).
// ---------------------------------------------------------------------------
__global__ __launch_bounds__(256) void gate_topk_kernel(
    const float* __restrict__ raw, const float* __restrict__ noise_z,
    float* __restrict__ gates) {
  const int wave = threadIdx.x >> 5, lane = threadIdx.x & 31;
  const int row = blockIdx.x * 8 + wave;
  if (row >= NB) return;

  float cand[3];
#pragma unroll
  for (int q = 0; q < 3; ++q) {
    int p = lane + q * 32;
    if (p < NP) {
      float clean = raw[row * 176 + p];
      float nr = raw[row * 176 + NP + p];
      float sp = (nr > 20.f) ? nr : log1pf(expf(nr));
      cand[q] = clean + noise_z[row * NP + p] * (sp + 0.01f);
      gates[row * NP + p] = 0.f;
    } else {
      cand[q] = -3.4e38f;
    }
  }

  float topv[3];
  int topi[3];
#pragma unroll
  for (int sel = 0; sel < 3; ++sel) {
    float mv = -3.4e38f;
    int mi = 0x7fffffff;
#pragma unroll
    for (int q = 0; q < 3; ++q) {
      int p = lane + q * 32;
      if (cand[q] > mv || (cand[q] == mv && p < mi)) { mv = cand[q]; mi = p; }
    }
    for (int off = 16; off > 0; off >>= 1) {
      float ov = __shfl_xor(mv, off);
      int oi = __shfl_xor(mi, off);
      if (ov > mv || (ov == mv && oi < mi)) { mv = ov; mi = oi; }
    }
    topv[sel] = mv;
    topi[sel] = mi;
    if ((mi & 31) == lane) cand[mi >> 5] = -3.4e38f;  // retire winner
  }

  if (lane == 0) {
    float m = topv[0];
    float e0 = expf(topv[0] - m), e1 = expf(topv[1] - m), e2 = expf(topv[2] - m);
    float s = e0 + e1 + e2;
    gates[row * NP + topi[0]] = e0 / s;
    gates[row * NP + topi[1]] = e1 / s;
    gates[row * NP + topi[2]] = e2 / s;
  }
}

// ---------------------------------------------------------------------------
// Host launcher
// ---------------------------------------------------------------------------
extern "C" void kernel_launch(void* const* d_in, const int* in_sizes, int n_in,
                              void* d_out, int out_size, void* d_ws, size_t ws_size,
                              hipStream_t stream) {
  (void)in_sizes; (void)n_in; (void)out_size; (void)ws_size;

  const float* x       = (const float*)d_in[0];
  const float* w_start = (const float*)d_in[1];
  const float* b_start = (const float*)d_in[2];
  const float* w1      = (const float*)d_in[3];
  const float* b1      = (const float*)d_in[4];   // flat = concat(b1[0], b1[1]) = bias1[2H]
  const float* w2      = (const float*)d_in[5];
  const float* b2      = (const float*)d_in[6];   // flat = bias2[2F]
  const float* w_gate  = (const float*)d_in[7];
  const float* w_noise = (const float*)d_in[8];
  const float* noise_z = (const float*)d_in[9];
  float* gates = (float*)d_out;

  char* ws = (char*)d_ws;
  __bf16* W1t  = (__bf16*)ws; ws += (size_t)N2H * N2F * 2;   // 32 MB
  __bf16* W2t  = (__bf16*)ws; ws += (size_t)N2F * N2H * 2;   // 32 MB
  __bf16* act1 = (__bf16*)ws; ws += (size_t)NB * N2F * 2;    //  8 MB
  __bf16* act2 = (__bf16*)ws; ws += (size_t)NB * N2H * 2;    // 32 MB
  float*  o2   = (float*)ws;  ws += (size_t)NB * N2F * 4;    // 16 MB
  float*  ampb = (float*)ws;  ws += (size_t)NB * NF * 4;     //  8 MB
  float*  raw  = (float*)ws;  ws += (size_t)NB * 176 * 4;    // 1.4 MB

  prep_w1t<<<(N2H * N2F) / 256, 256, 0, stream>>>(w1, W1t);
  prep_w2t<<<(N2F * N2H) / 256, 256, 0, stream>>>(w2, W2t);
  start_fft_kernel<<<NB, 256, 0, stream>>>(x, w_start, b_start, act1);
  // layer1: M=2048, N=8192, K=2048, relu -> bf16
  gemm_bf16<0><<<dim3(N2H / 128, NB / 128), 256, 0, stream>>>(
      act1, W1t, b1, (void*)act2, NB, N2H, N2F);
  // layer2: M=2048, N=2048, K=8192 -> f32
  gemm_bf16<1><<<dim3(N2F / 128, NB / 128), 256, 0, stream>>>(
      act2, W2t, b2, (void*)o2, NB, N2F, N2H);
  amp_kernel<<<(NB * NF) / 256, 256, 0, stream>>>(o2, ampb);
  router_kernel<<<NB / 16, 256, 0, stream>>>(ampb, w_gate, w_noise, raw);
  gate_topk_kernel<<<NB / 8, 256, 0, stream>>>(raw, noise_z, gates);
}
